// RNNTLoss_31009663877366
// MI455X (gfx1250) — compile-verified, matrix operand order checked by reading, not run on hardware
//
#include <hip/hip_runtime.h>
#include <hip/hip_bf16.h>
#include <math.h>

typedef _Float16 v16h __attribute__((ext_vector_type(16)));
typedef _Float16 v8h  __attribute__((ext_vector_type(8)));
typedef float    v8f  __attribute__((ext_vector_type(8)));

#define B_  8
#define T_  128
#define U_  64
#define U1_ 65
#define UP_ 80      // U1 padded to a multiple of 16 for WMMA tiles
#define V_  512

// ---------------------------------------------------------------------------
// Kernel 1: per-row max + exp for trans_acts -> f16 A matrix + f32 row max
// ---------------------------------------------------------------------------
__global__ void rowmax_exp_trans(const float* __restrict__ x,
                                 _Float16* __restrict__ a,
                                 float* __restrict__ mrow) {
    __shared__ float red[128];
    const int row = blockIdx.x;                 // 0 .. B*T-1
    const int tid = threadIdx.x;
    const float* xr = x + (size_t)row * V_;

    float m = -3.4e38f;
    for (int v = tid; v < V_; v += 128) m = fmaxf(m, xr[v]);
    red[tid] = m;
    __syncthreads();
    for (int s = 64; s > 0; s >>= 1) {
        if (tid < s) red[tid] = fmaxf(red[tid], red[tid + s]);
        __syncthreads();
    }
    m = red[0];

    _Float16* ar = a + (size_t)row * V_;
    for (int v = tid; v < V_; v += 128)
        ar[v] = (_Float16)__expf(xr[v] - m);
    if (tid == 0) mrow[row] = m;
}

// ---------------------------------------------------------------------------
// Kernel 2: same for pred_acts, padded to UP_ rows (pad rows are zero -> they
// contribute 0 to the GEMM and are masked in the epilogue anyway)
// ---------------------------------------------------------------------------
__global__ void rowmax_exp_pred(const float* __restrict__ x,
                                _Float16* __restrict__ bm,
                                float* __restrict__ mrow) {
    __shared__ float red[128];
    const int b   = blockIdx.x / UP_;
    const int u   = blockIdx.x % UP_;
    const int tid = threadIdx.x;
    _Float16* br = bm + ((size_t)b * UP_ + u) * V_;

    if (u >= U1_) {
        for (int v = tid; v < V_; v += 128) br[v] = (_Float16)0.0f;
        if (tid == 0) mrow[b * UP_ + u] = 0.0f;
        return;
    }
    const float* xr = x + ((size_t)b * U1_ + u) * V_;
    float m = -3.4e38f;
    for (int v = tid; v < V_; v += 128) m = fmaxf(m, xr[v]);
    red[tid] = m;
    __syncthreads();
    for (int s = 64; s > 0; s >>= 1) {
        if (tid < s) red[tid] = fmaxf(red[tid], red[tid + s]);
        __syncthreads();
    }
    m = red[0];
    for (int v = tid; v < V_; v += 128)
        br[v] = (_Float16)__expf(xr[v] - m);
    if (tid == 0) mrow[b * UP_ + u] = m;
}

// ---------------------------------------------------------------------------
// Kernel 3: one wave per 16x16 (t,u) tile. Z = A x B^T via WMMA f16->f32,
// K = 512 in 16 chunks of 32. Epilogue: lse = mt+mu+log(Z); emit lp_blank and
// lp_lab (gather trans/pred at vocab index 0 and labels[b,u]).
//
// 16-bit operand layout (ISA 7.12.2): lane l -> row/col = l&15, half = l>>4;
// vector elems e<8 cover K = kc + half*8 + e (contiguous 16B), elems e>=8
// cover K = kc + 16 + half*8 + (e-8). Identical map for A rows and B columns.
// ---------------------------------------------------------------------------
__global__ void joint_lse_wmma(const _Float16* __restrict__ A,
                               const _Float16* __restrict__ Bm,
                               const float* __restrict__ mt,
                               const float* __restrict__ mu,
                               const float* __restrict__ trans,
                               const float* __restrict__ pred,
                               const int*   __restrict__ labels,
                               float* __restrict__ lp_blank,
                               float* __restrict__ lp_lab) {
    const int tile = blockIdx.x;                // b*40 + tt*5 + uu
    const int b    = tile / 40;
    const int r    = tile % 40;
    const int t0   = (r / 5) * 16;
    const int u0   = (r % 5) * 16;

    const int l    = threadIdx.x;               // 0..31 (one wave)
    const int mr   = l & 15;
    const int half = l >> 4;

    const _Float16* arow = A  + ((size_t)(b * T_  + t0 + mr)) * V_ + half * 8;
    const _Float16* brow = Bm + ((size_t)(b * UP_ + u0 + mr)) * V_ + half * 8;

    v8f c = {};
    for (int kc = 0; kc < V_; kc += 32) {
        union { v16h v; v8h h[2]; } av, bv;
        av.h[0] = *(const v8h*)(arow + kc);
        av.h[1] = *(const v8h*)(arow + kc + 16);
        bv.h[0] = *(const v8h*)(brow + kc);
        bv.h[1] = *(const v8h*)(brow + kc + 16);
        c = __builtin_amdgcn_wmma_f32_16x16x32_f16(
                /*neg_a=*/false, av.v, /*neg_b=*/false, bv.v,
                /*c_mod=*/(short)0, c, /*reuse_a=*/false, /*reuse_b=*/false);
    }

    // D layout: lanes 0-15 -> M = rr,   N = lane
    //           lanes 16-31-> M = 8+rr, N = lane-16
    const int u = u0 + mr;
    if (u >= U1_) return;
    const float muv  = mu[b * UP_ + u];
    const float p0   = pred[((size_t)b * U1_ + u) * V_];             // pred[b,u,0]
    const int   lab  = (u < U_) ? labels[b * U_ + u] : 0;
    const float plab = (u < U_) ? pred[((size_t)b * U1_ + u) * V_ + lab] : 0.0f;

    for (int rr = 0; rr < 8; ++rr) {
        const int t = t0 + half * 8 + rr;
        const size_t tb = (size_t)(b * T_ + t);
        const float lse = mt[b * T_ + t] + muv + __logf(c[rr]);
        lp_blank[tb * U1_ + u] = trans[tb * V_] + p0 - lse;
        if (u < U_)
            lp_lab[tb * U_ + u] = trans[tb * V_ + lab] + plab - lse;
    }
}

// ---------------------------------------------------------------------------
// Kernel 4: alpha DP, anti-diagonal wavefront. One block per batch, alpha in
// LDS (128*65 floats = 33 KB of the 320 KB WGP pool). 192 barriered steps.
// ---------------------------------------------------------------------------
__global__ void alpha_wavefront(const float* __restrict__ lp_blank,
                                const float* __restrict__ lp_lab,
                                const int*   __restrict__ act_lens,
                                const int*   __restrict__ label_lens,
                                float* __restrict__ costs) {
    __shared__ float alpha[T_][U1_];
    const int b = blockIdx.x;
    const int u = threadIdx.x;                  // active for u < U1_
    const float* lpb = lp_blank + (size_t)b * T_ * U1_;
    const float* lpl = lp_lab   + (size_t)b * T_ * U_;

    for (int d = 0; d <= (T_ - 1) + (U1_ - 1); ++d) {
        const int t = d - u;
        if (u < U1_ && t >= 0 && t < T_) {
            float val;
            if (t == 0 && u == 0) {
                val = 0.0f;
            } else if (t == 0) {
                val = alpha[0][u - 1] + lpl[u - 1];
            } else if (u == 0) {
                val = alpha[t - 1][0] + lpb[(size_t)(t - 1) * U1_];
            } else {
                const float x = alpha[t - 1][u] + lpb[(size_t)(t - 1) * U1_ + u];
                const float y = alpha[t][u - 1] + lpl[(size_t)t * U_ + (u - 1)];
                const float mx = fmaxf(x, y), mn = fminf(x, y);
                val = mx + log1pf(__expf(mn - mx));
            }
            alpha[t][u] = val;
        }
        __syncthreads();
    }

    if (u == 0) {
        const int ti = act_lens[b] - 1;
        const int ui = label_lens[b];
        costs[b] = -(alpha[ti][ui] + lpb[(size_t)ti * U1_ + ui]);
    }
}

// ---------------------------------------------------------------------------
// Kernel 5: deterministic scalar sum of the 8 per-batch costs.
// ---------------------------------------------------------------------------
__global__ void sum_costs(const float* __restrict__ costs, float* __restrict__ out) {
    if (threadIdx.x == 0) {
        float s = 0.0f;
        for (int i = 0; i < B_; ++i) s += costs[i];
        out[0] = s;
    }
}

// ---------------------------------------------------------------------------
extern "C" void kernel_launch(void* const* d_in, const int* in_sizes, int n_in,
                              void* d_out, int out_size, void* d_ws, size_t ws_size,
                              hipStream_t stream) {
    (void)in_sizes; (void)n_in; (void)out_size; (void)ws_size;
    const float* trans    = (const float*)d_in[0];   // (B,T,V)  f32
    const float* pred     = (const float*)d_in[1];   // (B,U1,V) f32
    const int*   labels   = (const int*)d_in[2];     // (B,U)    i32
    const int*   act_lens = (const int*)d_in[3];     // (B,)     i32
    const int*   lab_lens = (const int*)d_in[4];     // (B,)     i32

    char* ws = (char*)d_ws;
    size_t off = 0;
    auto carve = [&](size_t bytes) {
        char* p = ws + off;
        off = (off + bytes + 255) & ~(size_t)255;
        return p;
    };
    _Float16* Af   = (_Float16*)carve((size_t)B_ * T_  * V_ * sizeof(_Float16)); // 1.0 MB
    _Float16* Bf   = (_Float16*)carve((size_t)B_ * UP_ * V_ * sizeof(_Float16)); // 0.64 MB
    float*    mt   = (float*)   carve((size_t)B_ * T_  * sizeof(float));
    float*    mu   = (float*)   carve((size_t)B_ * UP_ * sizeof(float));
    float*    lpb  = (float*)   carve((size_t)B_ * T_ * U1_ * sizeof(float));    // 0.27 MB
    float*    lpl  = (float*)   carve((size_t)B_ * T_ * U_  * sizeof(float));    // 0.26 MB
    float*    cost = (float*)   carve((size_t)B_ * sizeof(float));

    rowmax_exp_trans<<<B_ * T_,  128, 0, stream>>>(trans, Af, mt);
    rowmax_exp_pred <<<B_ * UP_, 128, 0, stream>>>(pred,  Bf, mu);
    joint_lse_wmma  <<<B_ * 8 * 5, 32, 0, stream>>>(Af, Bf, mt, mu, trans, pred,
                                                    labels, lpb, lpl);
    alpha_wavefront <<<B_, 128, 0, stream>>>(lpb, lpl, act_lens, lab_lens, cost);
    sum_costs       <<<1, 32, 0, stream>>>(cost, (float*)d_out);
}
